// SGFA_46213848105623
// MI455X (gfx1250) — matrix operand order repaired and checked by reading, not compiled, not run on hardware
//
#include <hip/hip_runtime.h>
#include <hip/hip_bf16.h>
#include <stdint.h>

// ---------------------------------------------------------------------------
// Contextual attention on MI455X (gfx1250), bf16 WMMA flash-attention.
//
// Per sample b:  Q[4096,576] = 3x3xC fg patches,  Kn = Q row-normalized,
//                Vt[1024,4096] = bg 4x4xC patches stored value-dim-major,
//   S[l,q] = Kn[l]·Q[q],  attn = softmax_l(10*S),  O[q,:] = sum_l attn * V[l,:]
//   scatter O as overlapping stride-2 4x4 tiles, /4.
// 215 GFLOP total vs ~32MB of HBM I/O -> matrix-unit bound: everything runs
// through v_wmma_f32_16x16x32_bf16; the 4096x4096 score matrix is never
// materialized (online softmax, score tiles live in accumulators/LDS).
// NOTE: reference clips softmax to >=1e-8 before PV; that perturbs the result
// by <= 4096*1e-8 relative mass (~4e-5) and is intentionally folded out.
// ---------------------------------------------------------------------------

typedef __bf16 bf16;
typedef __attribute__((ext_vector_type(16))) __bf16 v16bf;
typedef __attribute__((ext_vector_type(8)))  float  v8f;

union Frag { uint4 u[2]; v16bf v; };

#define B_   4
#define C_   64
#define H_   128
#define W_   128
#define HS   64
#define WS   64
#define L_   4096          // HS*WS query/key positions
#define KF   576           // C_*9 feature dim  (18 k-steps of 32)
#define VD   1024          // C_*16 value dim
#define QSTR 584           // LDS row stride (bf16) for Q tile (conflict-free)
#define PSTR 136           // LDS row stride (bf16) for P tile (conflict-free)

// workspace layout (bytes); total ~75.5 MB
#define OFF_FGS 0
#define SZ_FGS  (B_*C_*HS*WS*4)
#define OFF_Q   (OFF_FGS + SZ_FGS)
#define SZ_QK   (B_*L_*KF*2)
#define OFF_KN  (OFF_Q + SZ_QK)
#define OFF_VT  (OFF_KN + SZ_QK)

static __device__ __forceinline__ float lane_bcast(float v, int srcLane) {
    return __int_as_float(__builtin_amdgcn_ds_bpermute(srcLane << 2, __float_as_int(v)));
}
static __device__ __forceinline__ float xor16(float v, int lane) {
    return __int_as_float(__builtin_amdgcn_ds_bpermute(((lane ^ 16) & 31) << 2, __float_as_int(v)));
}

// ---------------------------------------------------------------- zero output
__global__ void k_zero(float* __restrict__ out, int n) {
    int i = blockIdx.x * 256 + threadIdx.x;
    if (i < n) out[i] = 0.0f;
}

// --------------------------------------- bilinear align_corners 128 -> 64 fg
__global__ void k_downsample(const float* __restrict__ fg, float* __restrict__ fgs) {
    int idx = blockIdx.x * 256 + threadIdx.x;
    if (idx >= B_ * C_ * HS * WS) return;
    int x = idx & 63, y = (idx >> 6) & 63, c = (idx >> 12) & 63, b = idx >> 18;
    const float sc = 127.0f / 63.0f;
    float fy = y * sc, fx = x * sc;
    int y0 = (int)floorf(fy); int y1 = min(y0 + 1, 127); float wy = fy - (float)y0;
    int x0 = (int)floorf(fx); int x1 = min(x0 + 1, 127); float wx = fx - (float)x0;
    const float* p = fg + ((size_t)(b * C_ + c)) * (H_ * W_);
    float v = p[y0 * W_ + x0] * (1.f - wy) * (1.f - wx)
            + p[y1 * W_ + x0] * wy * (1.f - wx)
            + p[y0 * W_ + x1] * (1.f - wy) * wx
            + p[y1 * W_ + x1] * wy * wx;
    fgs[idx] = v;
}

// ------------------------------- Q (raw patches) + Kn (L2-normalized), bf16
// block = 64 threads (one per channel), grid = (L_, B_)
__global__ void k_build_qk(const float* __restrict__ fgs,
                           bf16* __restrict__ Q, bf16* __restrict__ Kn) {
    int l = blockIdx.x, b = blockIdx.y;
    int y = l >> 6, x = l & 63;
    int c = threadIdx.x;                         // 0..63
    const float* src = fgs + ((size_t)(b * C_ + c)) * (HS * WS);
    float vals[9]; float ss = 0.f;
#pragma unroll
    for (int r = 0; r < 9; ++r) {
        int ky = r / 3, kx = r % 3;
        int yy = y + ky - 1, xx = x + kx - 1;
        float v = (yy >= 0 && yy < HS && xx >= 0 && xx < WS) ? src[yy * WS + xx] : 0.f;
        vals[r] = v; ss += v * v;
    }
    __shared__ float red[64];
    __shared__ float inv_norm;
    red[c] = ss; __syncthreads();
    if (c < 32) red[c] += red[c + 32];
    __syncthreads();
    if (c == 0) {
        float t = 0.f;
#pragma unroll
        for (int i = 0; i < 32; ++i) t += red[i];
        inv_norm = 1.0f / fmaxf(sqrtf(t), 1e-4f);
    }
    __syncthreads();
    float inv = inv_norm;
    size_t base = ((size_t)b * L_ + l) * KF + c * 9;
#pragma unroll
    for (int r = 0; r < 9; ++r) {
        Q [base + r] = (bf16)vals[r];
        Kn[base + r] = (bf16)(vals[r] * inv);
    }
}

// -------------------- Vt[b][v=c*16+ky*4+kx][l] = bg[b][c][2y+ky-1][2x+kx-1]
__global__ void k_build_vt(const float* __restrict__ bg, bf16* __restrict__ Vt) {
    int idx = blockIdx.x * 256 + threadIdx.x;       // == ((b*VD+v)*L_ + l)
    if (idx >= B_ * VD * L_) return;
    int l = idx & (L_ - 1);
    int v = (idx >> 12) & (VD - 1);
    int b = idx >> 22;
    int cch = v >> 4, ky = (v >> 2) & 3, kx = v & 3;
    int y = l >> 6, x = l & 63;
    int Y = 2 * y + ky - 1, X = 2 * x + kx - 1;
    float val = (Y >= 0 && Y < H_ && X >= 0 && X < W_)
                ? bg[(((size_t)b * C_ + cch) * H_ + Y) * W_ + X] : 0.f;
    Vt[idx] = (bf16)val;
}

// ---------------------------------------------------------- flash attention
// grid (L_/16, B_), block 256 (8 waves). Wave w: one 16x16 S^T tile per key
// block (M=keys -> softmax stats are per-lane), and 128 V columns for PV.
__global__ __launch_bounds__(256)
void k_attn(const bf16* __restrict__ Qg, const bf16* __restrict__ Kng,
            const bf16* __restrict__ Vtg, float* __restrict__ out) {
    const int b    = blockIdx.y;
    const int qt   = blockIdx.x;
    const int tid  = threadIdx.x;
    const int w    = tid >> 5;
    const int lane = tid & 31;
    const int half = lane >> 4;
    const int rm   = lane & 15;

    __shared__ __align__(16) bf16 Qlds[16 * QSTR];
    __shared__ __align__(16) bf16 Plds[16 * PSTR];
    __shared__ float redmax[8][16];
    __shared__ float redsum[8][16];

    // stage 16x576 bf16 Q tile into LDS (padded stride for bank-free reads)
    {
        const unsigned* qsrc =
            (const unsigned*)(Qg + ((size_t)b * L_ + (size_t)qt * 16) * KF);
        for (int i = tid; i < 16 * (KF / 2); i += 256) {
            int qq = i / (KF / 2), d = i % (KF / 2);
            ((unsigned*)(Qlds + qq * QSTR))[d] = qsrc[qq * (KF / 2) + d];
        }
    }
    __syncthreads();

    const v8f ZERO8 = {0.f, 0.f, 0.f, 0.f, 0.f, 0.f, 0.f, 0.f};
    v8f accO[8];
#pragma unroll
    for (int nt = 0; nt < 8; ++nt) accO[nt] = ZERO8;
    float m_run = -1e30f, s_run = 0.f;

    const bf16* KnB = Kng + (size_t)b * L_ * KF;
    const bf16* VtB = Vtg + (size_t)b * VD * L_;
    const float SM_C = 10.0f * 1.44269504088896340736f;   // scale * log2(e)

    for (int kb = 0; kb < L_ / 128; ++kb) {               // 32 key blocks
        const int l0 = kb * 128 + w * 16;
        // ---- S^T tile: A = Kn rows (M=16 keys), B = Q^T from LDS (N=16 q)
        v8f accS = ZERO8;
#pragma unroll
        for (int ks = 0; ks < KF / 32; ++ks) {
            Frag a, bb;
            const bf16* ap = KnB + (size_t)(l0 + rm) * KF + ks * 32 + half * 8;
            a.u[0] = *(const uint4*)ap;
            a.u[1] = *(const uint4*)(ap + 16);
            const bf16* bp = Qlds + rm * QSTR + ks * 32 + half * 8;
            bb.u[0] = *(const uint4*)bp;
            bb.u[1] = *(const uint4*)(bp + 16);
            accS = __builtin_amdgcn_wmma_f32_16x16x32_bf16(
                false, a.v, false, bb.v, (short)0, accS, false, false);
        }
        // ---- online softmax (per-lane q = rm; rows l across VGPRs + halves)
        float z[8]; float tmax = -1e30f;
#pragma unroll
        for (int r = 0; r < 8; ++r) { z[r] = accS[r] * SM_C; tmax = fmaxf(tmax, z[r]); }
        tmax = fmaxf(tmax, xor16(tmax, lane));
        if (half == 0) redmax[w][rm] = tmax;
        __syncthreads();
        float m_new = m_run;
#pragma unroll
        for (int ww = 0; ww < 8; ++ww) m_new = fmaxf(m_new, redmax[ww][rm]);
        float alpha = exp2f(m_run - m_new);
        float al[8];
#pragma unroll
        for (int r = 0; r < 8; ++r) al[r] = lane_bcast(alpha, r + 8 * half);
#pragma unroll
        for (int nt = 0; nt < 8; ++nt)
#pragma unroll
            for (int r = 0; r < 8; ++r) accO[nt][r] *= al[r];
        float psum = 0.f;
#pragma unroll
        for (int r = 0; r < 8; ++r) {
            float p = exp2f(z[r] - m_new);
            psum += p;
            Plds[rm * PSTR + w * 16 + r + 8 * half] = (bf16)p;
        }
        psum += xor16(psum, lane);
        if (half == 0) redsum[w][rm] = psum;
        s_run *= alpha; m_run = m_new;
        __syncthreads();
#pragma unroll
        for (int ww = 0; ww < 8; ++ww) s_run += redsum[ww][rm];
        // ---- PV: A = P tile from LDS (M=16 q), B = Vt columns (N=16 v)
        Frag pa[4];
#pragma unroll
        for (int ksl = 0; ksl < 4; ++ksl) {
            const bf16* pp = Plds + rm * PSTR + ksl * 32 + half * 8;
            pa[ksl].u[0] = *(const uint4*)pp;
            pa[ksl].u[1] = *(const uint4*)(pp + 16);
        }
#pragma unroll
        for (int nt = 0; nt < 8; ++nt) {
            const bf16* vcol = VtB + (size_t)(w * 128 + nt * 16 + rm) * L_
                             + kb * 128 + half * 8;
#pragma unroll
            for (int ksl = 0; ksl < 4; ++ksl) {
                Frag vb;
                const bf16* vp = vcol + ksl * 32;
                vb.u[0] = *(const uint4*)vp;
                vb.u[1] = *(const uint4*)(vp + 16);
                accO[nt] = __builtin_amdgcn_wmma_f32_16x16x32_bf16(
                    false, pa[ksl].v, false, vb.v, (short)0, accO[nt], false, false);
            }
        }
    }

    // ---- finalize: /sum, /4, scatter-add overlapping stride-2 4x4 patches
    float sinv = 1.0f / s_run;
    float sr[8];
#pragma unroll
    for (int r = 0; r < 8; ++r) sr[r] = lane_bcast(sinv, r + 8 * half) * 0.25f;
#pragma unroll
    for (int nt = 0; nt < 8; ++nt) {
        int v = w * 128 + nt * 16 + rm;
        int cch = v >> 4, ky = (v >> 2) & 3, kx = v & 3;
#pragma unroll
        for (int r = 0; r < 8; ++r) {
            int qg = qt * 16 + r + 8 * half;
            int y = qg >> 6, x = qg & 63;
            int Y = 2 * y + ky - 1, X = 2 * x + kx - 1;
            if (Y >= 0 && Y < H_ && X >= 0 && X < W_) {
                atomicAdd(out + (((size_t)b * C_ + cch) * H_ + Y) * W_ + X,
                          accO[nt][r] * sr[r]);
            }
        }
    }
}

// ---------------------------------------------------------------------------
extern "C" void kernel_launch(void* const* d_in, const int* in_sizes, int n_in,
                              void* d_out, int out_size, void* d_ws, size_t ws_size,
                              hipStream_t stream) {
    const float* bg = (const float*)d_in[0];   // background [4,64,128,128]
    const float* fg = (const float*)d_in[1];   // foreground [4,64,128,128]
    float* out = (float*)d_out;                // [4,64,128,128]
    char* ws = (char*)d_ws;
    float* fgs = (float*)(ws + OFF_FGS);
    bf16*  Q   = (bf16*)(ws + OFF_Q);
    bf16*  Kn  = (bf16*)(ws + OFF_KN);
    bf16*  Vt  = (bf16*)(ws + OFF_VT);

    const int nout = B_ * C_ * H_ * W_;
    k_zero<<<dim3((nout + 255) / 256), dim3(256), 0, stream>>>(out, nout);
    k_downsample<<<dim3((B_ * C_ * HS * WS + 255) / 256), dim3(256), 0, stream>>>(fg, fgs);
    k_build_qk<<<dim3(L_, B_), dim3(64), 0, stream>>>(fgs, Q, Kn);
    k_build_vt<<<dim3((B_ * VD * L_) / 256), dim3(256), 0, stream>>>(bg, Vt);
    k_attn<<<dim3(L_ / 16, B_), dim3(256), 0, stream>>>(Q, Kn, Vt, out);
}
